// Multihead_Attention_59974923321714
// MI455X (gfx1250) — matrix operand order repaired and checked by reading, not compiled, not run on hardware
//
#include <hip/hip_runtime.h>
#include <hip/hip_bf16.h>
#include <cstdint>

typedef __attribute__((ext_vector_type(16))) __bf16 v16bf;
typedef __attribute__((ext_vector_type(8)))  float  v8f;
typedef __attribute__((ext_vector_type(4)))  unsigned int u32x4;
typedef __attribute__((ext_vector_type(8)))  int  i32x8;
typedef __attribute__((ext_vector_type(4)))  int  i32x4;

#define HIDDEN 1024
#define HEADS  16
#define HD     64
#define SEQ    2048
#define BATCH  2
#define LOG2E  1.44269504088896340736f

#if __has_builtin(__builtin_amdgcn_tensor_load_to_lds)
#define ATHENA_HAS_TDM 1
#endif

// Fragment register layouts (CDNA5 ISA 7.12.2, wave32, 16x16x32 bf16):
//  A (16x32 MxK): lane L: row = L&15, kbase = (L>>4)*8. a[0..7]=K kbase.., a[8..15]=K kbase+16..
//  B (32x16 KxN): lane L: col = L&15, kbase = (L>>4)*8. b[0..7]=K kbase.., b[8..15]=K kbase+16.. (same as A).
//  C/D (f32):     lane L: col = L&15, VGPR v holds row v + 8*(L>>4).
union FragBF { v16bf v; int4 q[2]; };
union Pack4  { int2 d; __bf16 h[4]; };

#ifdef ATHENA_HAS_TDM
// Pre-built TDM 2D descriptor; only global-address words change per issue.
struct Tdm2D { u32x4 g0; i32x8 g1; };

__device__ __forceinline__ void tdm_init(
    Tdm2D& t, unsigned lds_addr,
    unsigned td0, unsigned td1, unsigned tile0, unsigned tile1, unsigned stride0)
{
    t.g0[0] = 1u;                          // count=1 (valid user D#)
    t.g0[1] = lds_addr;                    // LDS byte address
    t.g0[2] = 0u;
    t.g0[3] = 0u;
    t.g1[0] = (int)((1u << 16)             // data_size = 1 (2 bytes)
            | (1u << 20)                   // pad_enable
            | (4u << 22) | (7u << 25));    // 32 dwords -> +8 dwords pad
    t.g1[1] = (int)((td0 & 0xffffu) << 16);
    t.g1[2] = (int)(((td0 >> 16) & 0xffffu) | ((td1 & 0xffffu) << 16));
    t.g1[3] = (int)(((td1 >> 16) & 0xffffu) | ((tile0 & 0xffffu) << 16));
    t.g1[4] = (int)(tile1 & 0xffffu);      // tile_dim2 = 0
    t.g1[5] = (int)stride0;
    t.g1[6] = 0;
    t.g1[7] = 0;
}

__device__ __forceinline__ void tdm_issue(Tdm2D& t, unsigned long long ga)
{
    t.g0[2] = (unsigned)ga;                             // global_addr[31:0]
    t.g0[3] = (unsigned)((ga >> 32) & 0x01ffffffu)      // global_addr[56:32]
            | 0x80000000u;                              // type = 2 ("image")
    i32x4 z4 = {0, 0, 0, 0};
#if __clang_major__ >= 23
    i32x8 z8 = {};
    __builtin_amdgcn_tensor_load_to_lds(t.g0, t.g1, z4, z4, z8, 0);
#else
    __builtin_amdgcn_tensor_load_to_lds(t.g0, t.g1, z4, z4, 0);
#endif
}
#endif

__device__ __forceinline__ void wait_tensorcnt0() {
#if __has_builtin(__builtin_amdgcn_s_wait_tensorcnt)
    __builtin_amdgcn_s_wait_tensorcnt(0);
#else
    asm volatile("s_wait_tensorcnt 0x0" ::: "memory");
#endif
}
__device__ __forceinline__ void wait_tensorcnt2() {
#if __has_builtin(__builtin_amdgcn_s_wait_tensorcnt)
    __builtin_amdgcn_s_wait_tensorcnt(2);
#else
    asm volatile("s_wait_tensorcnt 0x2" ::: "memory");
#endif
}

// ---------------------------------------------------------------------------
// QKV projection (register-prefetch pipelined GEMM).
// transposed=0: out[b][h][s][d]  (Q, K)
// transposed=1: out[b][h][d][s]  (V stored pre-transposed for attention)
// ---------------------------------------------------------------------------
#define PROJ_BM 128
#define PROJ_BN 64
#define PROJ_BK 32
#define A_LD  40
#define BT_LD 40

__global__ __launch_bounds__(256) void qkv_proj_kernel(
    const float* __restrict__ X, const float* __restrict__ W,
    const float* __restrict__ bias, __bf16* __restrict__ out, int transposed)
{
    __shared__ __bf16 At[PROJ_BM * A_LD];
    __shared__ __bf16 Bt[PROJ_BN * BT_LD];   // transposed [n][k]

    const int tid  = threadIdx.x;
    const int wave = tid >> 5;
    const int lane = tid & 31;
    const int l15  = lane & 15;
    const int hi   = lane >> 4;

    const int m0 = blockIdx.y * PROJ_BM;
    const int n0 = blockIdx.x * PROJ_BN;

    // Per-thread staging source pointers, advanced by constant stride per step.
    const float* xsrc[4];
    const float* wsrc[2];
    int xr[4], xc[4], wr[2], wc[2];
    #pragma unroll
    for (int i = 0; i < 4; ++i) {
        int idx = (tid + i * 256) * 4;
        xr[i] = idx >> 5; xc[i] = idx & 31;
        xsrc[i] = &X[(size_t)(m0 + xr[i]) * HIDDEN + xc[i]];
    }
    #pragma unroll
    for (int i = 0; i < 2; ++i) {
        int idx = (tid + i * 256) * 4;
        wr[i] = idx >> 6; wc[i] = idx & 63;
        wsrc[i] = &W[(size_t)wr[i] * HIDDEN + n0 + wc[i]];
    }

    v8f acc[4] = {};
    float4 xa[4], wb[2];

    #pragma unroll
    for (int i = 0; i < 4; ++i) xa[i] = *(const float4*)xsrc[i];
    #pragma unroll
    for (int i = 0; i < 2; ++i) wb[i] = *(const float4*)wsrc[i];

    for (int k0 = 0; k0 < HIDDEN; k0 += PROJ_BK) {
        __syncthreads();                       // LDS free (prev tile consumed)
        #pragma unroll
        for (int i = 0; i < 4; ++i) {
            Pack4 p;
            p.h[0] = (__bf16)xa[i].x; p.h[1] = (__bf16)xa[i].y;
            p.h[2] = (__bf16)xa[i].z; p.h[3] = (__bf16)xa[i].w;
            *(int2*)&At[xr[i] * A_LD + xc[i]] = p.d;
        }
        #pragma unroll
        for (int i = 0; i < 2; ++i) {
            Bt[(wc[i] + 0) * BT_LD + wr[i]] = (__bf16)wb[i].x;
            Bt[(wc[i] + 1) * BT_LD + wr[i]] = (__bf16)wb[i].y;
            Bt[(wc[i] + 2) * BT_LD + wr[i]] = (__bf16)wb[i].z;
            Bt[(wc[i] + 3) * BT_LD + wr[i]] = (__bf16)wb[i].w;
        }
        __syncthreads();                       // tile published
        if (k0 + PROJ_BK < HIDDEN) {           // overlap next loads with WMMAs
            #pragma unroll
            for (int i = 0; i < 4; ++i) {
                xsrc[i] += PROJ_BK;
                xa[i] = *(const float4*)xsrc[i];
            }
            #pragma unroll
            for (int i = 0; i < 2; ++i) {
                wsrc[i] += (size_t)PROJ_BK * HIDDEN;
                wb[i] = *(const float4*)wsrc[i];
            }
        }

        FragBF a;
        const __bf16* ap = &At[(wave * 16 + l15) * A_LD + hi * 8];
        a.q[0] = *(const int4*)(ap);
        a.q[1] = *(const int4*)(ap + 16);
        #pragma unroll
        for (int j = 0; j < 4; ++j) {
            FragBF bfr;
            const __bf16* bp = &Bt[(j * 16 + l15) * BT_LD + hi * 8];
            bfr.q[0] = *(const int4*)(bp);
            bfr.q[1] = *(const int4*)(bp + 16);
            acc[j] = __builtin_amdgcn_wmma_f32_16x16x32_bf16(
                false, a.v, false, bfr.v, (short)0, acc[j], false, false);
        }
    }

    #pragma unroll
    for (int j = 0; j < 4; ++j) {
        int n = n0 + j * 16 + l15;
        float bn = bias[n];
        int h = n >> 6, d = n & 63;
        #pragma unroll
        for (int v = 0; v < 8; ++v) {
            int m = m0 + wave * 16 + v + 8 * hi;
            int b = m >> 11, s = m & (SEQ - 1);
            __bf16 val = (__bf16)(acc[j][v] + bn);
            if (!transposed)
                out[((size_t)((b * HEADS + h) * SEQ + s)) * HD + d] = val;
            else
                out[((size_t)((b * HEADS + h) * HD + d)) * SEQ + s] = val;
        }
    }
}

// ---------------------------------------------------------------------------
// Flash attention. K ws layout [b][h][s][d]; V ws layout [b][h][d][s] (pre-T).
// TDM path: double-buffered K/V tiles; tensor loads for tile i+1 overlap
// compute of tile i (tensor ops complete in-order -> wait tensorcnt<=2).
// ---------------------------------------------------------------------------
#define ATT_BK 64
#define K_LD 80
#define V_LD 80
#define P_LD 72
#define KT_ELEMS (ATT_BK * K_LD)
#define VT_ELEMS (HD * V_LD)
#define N_ITERS  (SEQ / ATT_BK)

__global__ __launch_bounds__(256) void attn_kernel(
    const __bf16* __restrict__ Q, const __bf16* __restrict__ K,
    const __bf16* __restrict__ Vg, const float* __restrict__ mask,
    float* __restrict__ out)
{
#ifdef ATHENA_HAS_TDM
    __shared__ __bf16 Kt[2 * KT_ELEMS];      // double buffered [kv][d]
    __shared__ __bf16 Vt[2 * VT_ELEMS];      // double buffered [d][kv]
#else
    __shared__ __bf16 Kt[KT_ELEMS];
    __shared__ __bf16 Vt[VT_ELEMS];
#endif
    __shared__ __bf16 Pt[8][16 * P_LD];      // per-wave [q][kv]

    const int tid  = threadIdx.x;
    const int wave = tid >> 5;
    const int lane = tid & 31;
    const int l15  = lane & 15;
    const int hi   = lane >> 4;

    const int bh = blockIdx.y;
    const int b  = bh >> 4;
    const int h  = bh & (HEADS - 1);
    const int q0 = blockIdx.x * 128 + wave * 16;

    const __bf16* Qh = Q  + (size_t)bh * SEQ * HD;
    const __bf16* Kh = K  + (size_t)bh * SEQ * HD;
    const __bf16* Vh = Vg + (size_t)bh * HD * SEQ;   // [d][s]
    const float*  mrow = mask + (size_t)b * SEQ;

    FragBF qa[2];
    #pragma unroll
    for (int ks = 0; ks < 2; ++ks) {
        const __bf16* qp = &Qh[(size_t)(q0 + l15) * HD + ks * 32 + hi * 8];
        qa[ks].q[0] = *(const int4*)(qp);
        qa[ks].q[1] = *(const int4*)(qp + 16);
    }

    v8f o[4] = {};
    float m_run[8], l_run[8];
    #pragma unroll
    for (int v = 0; v < 8; ++v) { m_run[v] = -3.0e38f; l_run[v] = 0.0f; }

    const float scale2 = 0.125f * LOG2E;     // QK^T scale folded with log2(e)

#ifdef ATHENA_HAS_TDM
    // Descriptors built ONCE; per issue only global-address words change.
    Tdm2D dK[2], dV[2];
    {
        unsigned ldsK = (unsigned)(uintptr_t)&Kt[0];
        unsigned ldsV = (unsigned)(uintptr_t)&Vt[0];
        tdm_init(dK[0], ldsK,                  HD,  SEQ, HD,     ATT_BK, HD);
        tdm_init(dK[1], ldsK + 2 * KT_ELEMS,   HD,  SEQ, HD,     ATT_BK, HD);
        tdm_init(dV[0], ldsV,                  SEQ, HD,  ATT_BK, HD,     SEQ);
        tdm_init(dV[1], ldsV + 2 * VT_ELEMS,   SEQ, HD,  ATT_BK, HD,     SEQ);
    }
    const unsigned long long gaK0 = (unsigned long long)(uintptr_t)Kh;
    const unsigned long long gaV0 = (unsigned long long)(uintptr_t)Vh;
    if (wave == 0) {
        tdm_issue(dK[0], gaK0);
        tdm_issue(dV[0], gaV0);
    }
#endif

    for (int it = 0; it < N_ITERS; ++it) {
        const int kv0 = it * ATT_BK;
#ifdef ATHENA_HAS_TDM
        const int buf = it & 1;
        __syncthreads();   // all waves done reading buf^1 (iter it-1)
        if (it + 1 < N_ITERS) {
            if (wave == 0) {
                tdm_issue(dK[buf ^ 1], gaK0 + (size_t)(kv0 + ATT_BK) * HD * 2);
                tdm_issue(dV[buf ^ 1], gaV0 + (size_t)(kv0 + ATT_BK) * 2);
                wait_tensorcnt2();   // tile `it` done (in-order); next 2 in flight
            }
        } else if (wave == 0) {
            wait_tensorcnt0();
        }
        __syncthreads();   // tile `it` published
        const __bf16* Kb = &Kt[buf * KT_ELEMS];
        const __bf16* Vb = &Vt[buf * VT_ELEMS];
#else
        __syncthreads();
        #pragma unroll
        for (int i = 0; i < 2; ++i) {
            int idx = (tid + i * 256) * 8;
            int r = idx >> 6, c = idx & 63;
            *(int4*)&Kt[r * K_LD + c] =
                *(const int4*)&Kh[(size_t)(kv0 + r) * HD + c];
            *(int4*)&Vt[r * V_LD + c] =
                *(const int4*)&Vh[(size_t)r * SEQ + kv0 + c];
        }
        __syncthreads();
        const __bf16* Kb = &Kt[0];
        const __bf16* Vb = &Vt[0];
#endif

        // ---- S = (Q K^T) * scale2 + mask*log2e (log2 domain) ------------
        v8f s[4] = {};
        #pragma unroll
        for (int j = 0; j < 4; ++j) {
            #pragma unroll
            for (int ks = 0; ks < 2; ++ks) {
                FragBF kb;
                const __bf16* kp = &Kb[(j * 16 + l15) * K_LD + ks * 32 + hi * 8];
                kb.q[0] = *(const int4*)(kp);
                kb.q[1] = *(const int4*)(kp + 16);
                s[j] = __builtin_amdgcn_wmma_f32_16x16x32_bf16(
                    false, qa[ks].v, false, kb.v, (short)0, s[j], false, false);
            }
        }
        #pragma unroll
        for (int j = 0; j < 4; ++j) {
            float mv = mrow[kv0 + j * 16 + l15] * LOG2E;
            #pragma unroll
            for (int v = 0; v < 8; ++v) s[j][v] = s[j][v] * scale2 + mv;
        }

        // ---- online softmax (base-2), C-frag layout (row = v + 8*hi) ----
        float pm[8];
        #pragma unroll
        for (int v = 0; v < 8; ++v)
            pm[v] = fmaxf(fmaxf(s[0][v], s[1][v]), fmaxf(s[2][v], s[3][v]));
        #pragma unroll
        for (int sh = 1; sh < 16; sh <<= 1)
            #pragma unroll
            for (int v = 0; v < 8; ++v)
                pm[v] = fmaxf(pm[v], __shfl_xor(pm[v], sh, 32));

        float alpha[8], rs[8];
        #pragma unroll
        for (int v = 0; v < 8; ++v) {
            float mn = fmaxf(m_run[v], pm[v]);
            alpha[v] = __builtin_amdgcn_exp2f(m_run[v] - mn);
            m_run[v] = mn;
            rs[v] = 0.0f;
        }
        #pragma unroll
        for (int j = 0; j < 4; ++j)
            #pragma unroll
            for (int v = 0; v < 8; ++v) {
                float p = __builtin_amdgcn_exp2f(s[j][v] - m_run[v]);
                s[j][v] = p;
                rs[v] += p;
            }
        #pragma unroll
        for (int sh = 1; sh < 16; sh <<= 1)
            #pragma unroll
            for (int v = 0; v < 8; ++v)
                rs[v] += __shfl_xor(rs[v], sh, 32);
        #pragma unroll
        for (int v = 0; v < 8; ++v)
            l_run[v] = l_run[v] * alpha[v] + rs[v];
        #pragma unroll
        for (int j = 0; j < 4; ++j)
            #pragma unroll
            for (int v = 0; v < 8; ++v)
                o[j][v] *= alpha[v];

        // ---- P -> per-wave LDS, reload as A fragments -------------------
        __bf16* pt = &Pt[wave][0];
        #pragma unroll
        for (int j = 0; j < 4; ++j)
            #pragma unroll
            for (int v = 0; v < 8; ++v)
                pt[(v + 8 * hi) * P_LD + j * 16 + l15] = (__bf16)s[j][v];
        asm volatile("s_wait_dscnt 0" ::: "memory");

        FragBF pa[2];
        #pragma unroll
        for (int ks = 0; ks < 2; ++ks) {
            const __bf16* pp = &pt[l15 * P_LD + ks * 32 + hi * 8];
            pa[ks].q[0] = *(const int4*)(pp);
            pa[ks].q[1] = *(const int4*)(pp + 16);
        }

        // ---- O += P @ V -------------------------------------------------
        #pragma unroll
        for (int j = 0; j < 4; ++j) {
            #pragma unroll
            for (int ks = 0; ks < 2; ++ks) {
                FragBF vb;
                const __bf16* vp = &Vb[(j * 16 + l15) * V_LD + ks * 32 + hi * 8];
                vb.q[0] = *(const int4*)(vp);
                vb.q[1] = *(const int4*)(vp + 16);
                o[j] = __builtin_amdgcn_wmma_f32_16x16x32_bf16(
                    false, pa[ks].v, false, vb.v, (short)0, o[j], false, false);
            }
        }
    }

    float inv_l[8];
    #pragma unroll
    for (int v = 0; v < 8; ++v) inv_l[v] = 1.0f / l_run[v];
    #pragma unroll
    for (int j = 0; j < 4; ++j)
        #pragma unroll
        for (int v = 0; v < 8; ++v) {
            int q = q0 + v + 8 * hi;
            int d = j * 16 + l15;
            out[((size_t)(b * SEQ + q)) * HIDDEN + h * HD + d] = o[j][v] * inv_l[v];
        }
}

// ---------------------------------------------------------------------------
extern "C" void kernel_launch(void* const* d_in, const int* in_sizes, int n_in,
                              void* d_out, int out_size, void* d_ws, size_t ws_size,
                              hipStream_t stream) {
    const float* former = (const float*)d_in[0];
    const float* latter = (const float*)d_in[1];
    const float* mask   = (const float*)d_in[2];
    const float* Wq = (const float*)d_in[3];
    const float* bq = (const float*)d_in[4];
    const float* Wk = (const float*)d_in[5];
    const float* bk = (const float*)d_in[6];
    const float* Wv = (const float*)d_in[7];
    const float* bv = (const float*)d_in[8];

    const size_t per = (size_t)BATCH * HEADS * SEQ * HD;  // 4M bf16 elems (8 MB)
    __bf16* qws = (__bf16*)d_ws;        // needs 3 * 8 MB = 24 MB of d_ws
    __bf16* kws = qws + per;
    __bf16* vws = kws + per;            // stored [b][h][d][s]

    dim3 pg(HIDDEN / PROJ_BN, (BATCH * SEQ) / PROJ_BM);   // 16 x 32
    qkv_proj_kernel<<<pg, 256, 0, stream>>>(former, Wq, bq, qws, 0);
    qkv_proj_kernel<<<pg, 256, 0, stream>>>(latter, Wk, bk, kws, 0);
    qkv_proj_kernel<<<pg, 256, 0, stream>>>(latter, Wv, bv, vws, 1);

    dim3 ag(SEQ / 128, BATCH * HEADS);                    // 16 x 32
    attn_kernel<<<ag, 256, 0, stream>>>(qws, kws, vws, mask, (float*)d_out);
}
